// AFAGNN_6889127543055
// MI455X (gfx1250) — matrix operand order repaired and checked by reading, not compiled, not run on hardware
//
#include <hip/hip_runtime.h>
#include <math.h>

// Problem constants (match reference)
#define N_NODES 100000
#define N_EDGES 1600000
#define F_IN    512
#define HID     64
#define NCLS    40
#define NSTRIPS (N_NODES / 16)   // 6250, exact

typedef __attribute__((ext_vector_type(16))) __bf16          v16bf;
typedef __attribute__((ext_vector_type(8)))  float           v8f;
typedef __attribute__((ext_vector_type(16))) unsigned short  ushort16;
typedef __attribute__((ext_vector_type(4)))  unsigned int    uint32x4;
typedef __attribute__((ext_vector_type(8)))  int             int32x8;
typedef __attribute__((ext_vector_type(4)))  int             int32x4;

static __device__ __forceinline__ unsigned short f2bf_rne(float f) {
  unsigned int u = __float_as_uint(f);
  unsigned int r = u + 0x7FFFu + ((u >> 16) & 1u);   // round-to-nearest-even
  return (unsigned short)(r >> 16);
}

// ---------------------------------------------------------------------------
// GEMM1: h0 = relu(x @ W1^T + b1)   [N,512] x [64,512]^T -> [N,64]
// bf16 WMMA, f32 accumulate. One wave per 16-row strip, 4 column tiles.
// W1 is DMA'd into LDS by the Tensor Data Mover (one 2D D# descriptor:
// 64x512 fp32 tensor == tile), then converted once to bf16 in LDS.
// ---------------------------------------------------------------------------
__global__ __launch_bounds__(256)
void gemm1_relu_wmma(const float* __restrict__ x,
                     const float* __restrict__ W1,
                     const float* __restrict__ b1,
                     float* __restrict__ h0) {
  __shared__ float          wf32[HID * F_IN];        // 128 KB: TDM target (fp32 W1)
  __shared__ unsigned short wbf [HID * F_IN];        //  64 KB: bf16 copy used by WMMA
  const int tid  = threadIdx.x;
  const int wave = tid >> 5;
  const int lane = tid & 31;

  // --- TDM: tensor_load_to_lds of the whole W1 (wave 0 issues one DMA) -----
  if (wave == 0) {
    unsigned long long ga = (unsigned long long)(size_t)W1;   // 57-bit VA
    unsigned int lds_off  = (unsigned int)(size_t)wf32;       // LDS byte offset

    uint32x4 g0;
    g0.x = 1u;                                  // count=1 (valid), user mode
    g0.y = lds_off;                             // D#.lds_addr
    g0.z = (unsigned int)(ga & 0xFFFFFFFFu);    // global_addr[31:0]
    g0.w = (unsigned int)((ga >> 32) & 0x01FFFFFFu) | (2u << 30); // addr[56:32] | type=2

    int32x8 g1;
    g1[0] = (int)(2u << 16);                    // wg_mask=0, data_size=2 (4 bytes)
    g1[1] = (int)((F_IN & 0xFFFFu) << 16);      // tensor_dim0[15:0] @ bits 63:48
    g1[2] = (int)((F_IN >> 16) | ((HID & 0xFFFFu) << 16)); // dim0 hi | dim1 lo
    g1[3] = (int)((HID >> 16) | ((unsigned)F_IN << 16));   // dim1 hi | tile_dim0=512
    g1[4] = (int)HID;                           // tile_dim1=64, tile_dim2=0
    g1[5] = (int)F_IN;                          // tensor_dim0_stride = 512 (low 32)
    g1[6] = 0;                                  // stride0 hi | stride1 lo (2D: unused)
    g1[7] = 0;

    int32x4 gz4; gz4[0] = 0; gz4[1] = 0; gz4[2] = 0; gz4[3] = 0;
    int32x8 gz8;
    for (int i = 0; i < 8; ++i) gz8[i] = 0;

    // clang-23 / therock-10.0 form: 6 args (extra int32x8 slot, zero for 2D)
    __builtin_amdgcn_tensor_load_to_lds(g0, g1, gz4, gz4, gz8, 0);
    __builtin_amdgcn_s_wait_tensorcnt(0);       // s_wait_tensorcnt 0
    asm volatile("" ::: "memory");              // LDS now holds W1 (keep LLVM honest)
  }
  __syncthreads();

  // one-shot fp32 -> bf16 conversion in LDS
  for (int i = tid; i < HID * F_IN; i += 256)
    wbf[i] = f2bf_rne(wf32[i]);
  __syncthreads();

  const int strip = blockIdx.x * 8 + wave;
  if (strip >= NSTRIPS) return;                 // wave-uniform: EXEC stays all-1s

  const int row0 = strip * 16;
  const int m    = lane & 15;                   // row (A) / column (B) within tile
  const int hl   = lane >> 4;                   // lane half select
  const float* xrow = x + (size_t)(row0 + m) * F_IN;

  v8f acc[4];
  for (int t = 0; t < 4; ++t)
    for (int j = 0; j < 8; ++j) acc[t][j] = 0.0f;

  for (int ks = 0; ks < F_IN / 32; ++ks) {
    const int K0 = ks * 32;
    __builtin_prefetch(xrow + K0 + 128, 0, 3);  // global_prefetch_b8, ~4 steps ahead

    // A fragment (16-bit A 16x32, contiguous k order matching B):
    //   e in [0,16): K = K0 + hl*16 + e
    float4 a0 = *(const float4*)(xrow + K0 + hl * 16);
    float4 a1 = *(const float4*)(xrow + K0 + hl * 16 + 4);
    float4 a2 = *(const float4*)(xrow + K0 + hl * 16 + 8);
    float4 a3 = *(const float4*)(xrow + K0 + hl * 16 + 12);
    ushort16 au;
    au[0]  = f2bf_rne(a0.x); au[1]  = f2bf_rne(a0.y);
    au[2]  = f2bf_rne(a0.z); au[3]  = f2bf_rne(a0.w);
    au[4]  = f2bf_rne(a1.x); au[5]  = f2bf_rne(a1.y);
    au[6]  = f2bf_rne(a1.z); au[7]  = f2bf_rne(a1.w);
    au[8]  = f2bf_rne(a2.x); au[9]  = f2bf_rne(a2.y);
    au[10] = f2bf_rne(a2.z); au[11] = f2bf_rne(a2.w);
    au[12] = f2bf_rne(a3.x); au[13] = f2bf_rne(a3.y);
    au[14] = f2bf_rne(a3.z); au[15] = f2bf_rne(a3.w);
    const v16bf afrag = __builtin_bit_cast(v16bf, au);

    for (int t = 0; t < 4; ++t) {
      // B fragment (32x16): lane holds column n=m of B = W1 row (t*16+m);
      // K = K0 + hl*16 + e, 16 contiguous bf16 -> two ds_load_b128
      const unsigned short* bp = &wbf[(size_t)(t * 16 + m) * F_IN + K0 + hl * 16];
      struct { uint4 lo, hi; } bb;
      bb.lo = *(const uint4*)(bp);
      bb.hi = *(const uint4*)(bp + 8);
      const v16bf bfrag = __builtin_bit_cast(v16bf, bb);

      acc[t] = __builtin_amdgcn_wmma_f32_16x16x32_bf16(
                 false, afrag, false, bfrag, (short)0, acc[t], false, false);
    }
  }

  // C/D layout: VGPR i, lane -> row = row0 + i + hl*8, col = t*16 + (lane&15)
  for (int t = 0; t < 4; ++t) {
    const int   col  = t * 16 + m;
    const float bias = b1[col];
    for (int i = 0; i < 8; ++i) {
      const int   r = row0 + i + hl * 8;
      const float v = acc[t][i] + bias;
      h0[(size_t)r * HID + col] = v > 0.0f ? v : 0.0f;
    }
  }
}

// ---------------------------------------------------------------------------
// Small helper kernels (edge phase is L2-resident on MI455X: h & agg < 192MB)
// ---------------------------------------------------------------------------
__global__ void zero_kernel(float* __restrict__ p, int n) {
  int i = blockIdx.x * blockDim.x + threadIdx.x;
  if (i < n) p[i] = 0.0f;
}

__global__ void degree_kernel(const int* __restrict__ row, float* __restrict__ deg) {
  int e = blockIdx.x * blockDim.x + threadIdx.x;
  if (e < N_EDGES) atomicAdd(&deg[row[e]], 1.0f);
}

__global__ void nd_kernel(float* __restrict__ deg_nd) {
  int i = blockIdx.x * blockDim.x + threadIdx.x;
  if (i < N_NODES) {
    float d = deg_nd[i];
    d = d < 1.0f ? 1.0f : d;
    deg_nd[i] = rsqrtf(d);
  }
}

// msg = (-nd[row]*nd[col]) * h_src[row]; agg[col] += msg    (16 lanes per edge)
__global__ void edge_agg_kernel(const int* __restrict__ row,
                                const int* __restrict__ col,
                                const float* __restrict__ nd,
                                const float* __restrict__ hsrc,
                                float* __restrict__ agg) {
  long long tid = (long long)blockIdx.x * blockDim.x + threadIdx.x;
  if (tid >= (long long)N_EDGES * 16) return;
  const int e = (int)(tid >> 4);
  const int f = ((int)tid & 15) * 4;
  const int r = row[e];
  const int c = col[e];
  const float g = -nd[r] * nd[c];
  const float4 hv = *(const float4*)(hsrc + (size_t)r * HID + f);
  float* ap = agg + (size_t)c * HID + f;
  atomicAdd(ap + 0, g * hv.x);
  atomicAdd(ap + 1, g * hv.y);
  atomicAdd(ap + 2, g * hv.z);
  atomicAdd(ap + 3, g * hv.w);
}

// h_dst = (sigmoid(eps1)+1) * raw + agg
__global__ void combine_kernel(const float* __restrict__ raw,
                               const float* __restrict__ agg,
                               const float* __restrict__ eps1,
                               float* __restrict__ hdst) {
  int i = blockIdx.x * blockDim.x + threadIdx.x;
  if (i >= N_NODES * HID) return;
  const int f = i & (HID - 1);
  const float s = 1.0f + 1.0f / (1.0f + __expf(-eps1[f]));
  hdst[i] = s * raw[i] + agg[i];
}

// ---------------------------------------------------------------------------
// Head: logits = h @ W2^T + b2 ; out = log_softmax(logits). One thread / row.
// All loops fully unrolled so hr[] / logits[] stay register-resident (no
// scratch spills); W2+b2 broadcast from LDS.
// ---------------------------------------------------------------------------
__global__ __launch_bounds__(256)
void head_kernel(const float* __restrict__ h,
                 const float* __restrict__ W2,
                 const float* __restrict__ b2,
                 float* __restrict__ out) {
  __shared__ float w2s[NCLS * HID];
  __shared__ float b2s[NCLS];
  for (int i = threadIdx.x; i < NCLS * HID; i += blockDim.x) w2s[i] = W2[i];
  if (threadIdx.x < NCLS) b2s[threadIdx.x] = b2[threadIdx.x];
  __syncthreads();

  const int r = blockIdx.x * blockDim.x + threadIdx.x;
  if (r >= N_NODES) return;

  float hr[HID];
  const float4* hp = (const float4*)(h + (size_t)r * HID);
#pragma unroll
  for (int i = 0; i < HID / 4; ++i) {
    float4 v = hp[i];
    hr[4 * i + 0] = v.x; hr[4 * i + 1] = v.y;
    hr[4 * i + 2] = v.z; hr[4 * i + 3] = v.w;
  }

  float logits[NCLS];
  float mx = -INFINITY;
#pragma unroll
  for (int c = 0; c < NCLS; ++c) {
    float s = b2s[c];
    const float* wr = &w2s[c * HID];
#pragma unroll
    for (int k = 0; k < HID; ++k) s += hr[k] * wr[k];
    logits[c] = s;
    mx = fmaxf(mx, s);
  }
  float sum = 0.0f;
#pragma unroll
  for (int c = 0; c < NCLS; ++c) sum += __expf(logits[c] - mx);
  const float lse = mx + __logf(sum);
  float* op = out + (size_t)r * NCLS;
#pragma unroll
  for (int c = 0; c < NCLS; ++c) op[c] = logits[c] - lse;
}

// ---------------------------------------------------------------------------
extern "C" void kernel_launch(void* const* d_in, const int* in_sizes, int n_in,
                              void* d_out, int out_size, void* d_ws, size_t ws_size,
                              hipStream_t stream) {
  const float* x    = (const float*)d_in[0];
  const int*   ei   = (const int*)  d_in[1];   // [2,E]: row then col
  const float* W1   = (const float*)d_in[2];
  const float* b1   = (const float*)d_in[3];
  // d_in[4..7] = gate weights/biases: dead code in the reference (g := -1)
  const float* eps1 = (const float*)d_in[8];
  const float* W2   = (const float*)d_in[9];
  const float* b2   = (const float*)d_in[10];

  const int* row = ei;
  const int* col = ei + N_EDGES;

  float* out_ls = (float*)d_out;                              // [N, 40]
  float* h_out  = (float*)d_out + (size_t)N_NODES * NCLS;     // [N, 64] (h after round 2)

  // workspace layout: nd | raw(h0) | h1 | agg  (~77 MB)
  float* nd  = (float*)d_ws;
  float* raw = nd  + N_NODES;
  float* h1  = raw + (size_t)N_NODES * HID;
  float* agg = h1  + (size_t)N_NODES * HID;

  const int NH = N_NODES * HID;

  // degrees -> nd
  zero_kernel  <<<(N_NODES + 255) / 256, 256, 0, stream>>>(nd, N_NODES);
  degree_kernel<<<(N_EDGES + 255) / 256, 256, 0, stream>>>(row, nd);
  nd_kernel    <<<(N_NODES + 255) / 256, 256, 0, stream>>>(nd);

  // big GEMM (TDM-staged W1 + bf16 WMMA): raw = relu(x @ W1^T + b1)
  gemm1_relu_wmma<<<(NSTRIPS + 7) / 8, 256, 0, stream>>>(x, W1, b1, raw);

  const long long nagg = (long long)N_EDGES * 16;
  const int agg_blocks = (int)((nagg + 255) / 256);

  // propagation round 1: raw -> h1
  zero_kernel    <<<(NH + 255) / 256, 256, 0, stream>>>(agg, NH);
  edge_agg_kernel<<<agg_blocks, 256, 0, stream>>>(row, col, nd, raw, agg);
  combine_kernel <<<(NH + 255) / 256, 256, 0, stream>>>(raw, agg, eps1, h1);

  // propagation round 2: h1 -> h_out (directly into d_out's h slot)
  zero_kernel    <<<(NH + 255) / 256, 256, 0, stream>>>(agg, NH);
  edge_agg_kernel<<<agg_blocks, 256, 0, stream>>>(row, col, nd, h1, agg);
  combine_kernel <<<(NH + 255) / 256, 256, 0, stream>>>(raw, agg, eps1, h_out);

  // classifier head + log_softmax
  head_kernel<<<(N_NODES + 255) / 256, 256, 0, stream>>>(h_out, W2, b2, out_ls);
}
